// GINE_60988535604051
// MI455X (gfx1250) — matrix operand order repaired and checked by reading, not compiled, not run on hardware
//
#include <hip/hip_runtime.h>

#define N_NODES_C   200000
#define N_EDGES_C   3200000
#define NUM_GRAPHS_C 1024
#define CH 64
#define NUM_LAYERS_C 4
#define LSTR 72   // padded LDS row stride in _Float16 units (avoids bank conflicts)

typedef __attribute__((ext_vector_type(16))) _Float16 v16h;
typedef __attribute__((ext_vector_type(8)))  float    v8f;

__device__ __forceinline__ v8f wmma_f32_f16(v16h a, v16h b, v8f c) {
  // 8 args: (neg_a, A, neg_b, B, c_mod, C, reuse_a, reuse_b)
  return __builtin_amdgcn_wmma_f32_16x16x32_f16(false, a, false, b, (short)0, c,
                                                false, false);
}

// ---------------- node embedding gather: h[n] = emb[x[n]] ----------------
__global__ __launch_bounds__(256) void k_embed(const int* __restrict__ x,
                                               const float* __restrict__ emb,
                                               float* __restrict__ h) {
  long gid = (long)blockIdx.x * blockDim.x + threadIdx.x;
  if (gid >= (long)N_NODES_C * 16) return;
  int node = (int)(gid >> 4), q = (int)(gid & 15);
  int t = x[node];
  reinterpret_cast<float4*>(h)[(long)node * 16 + q] =
      reinterpret_cast<const float4*>(emb)[(long)t * 16 + q];
}

// ---------------- zero fill (float4 granularity) ----------------
__global__ __launch_bounds__(256) void k_zero(float4* __restrict__ p, long n4) {
  long gid = (long)blockIdx.x * blockDim.x + threadIdx.x;
  if (gid < n4) p[gid] = make_float4(0.f, 0.f, 0.f, 0.f);
}

// ---------------- edge scatter: agg[dst] += h[src] ----------------
// one thread per (edge, 4-channel group); gathers hit L2 (h is 51 MB, L2 = 192 MB)
__global__ __launch_bounds__(256) void k_scatter(const int* __restrict__ ei,
                                                 const float* __restrict__ h,
                                                 float* __restrict__ agg) {
  long gid = (long)blockIdx.x * blockDim.x + threadIdx.x;
  if (gid >= (long)N_EDGES_C * 16) return;
  int e = (int)(gid >> 4), q = (int)(gid & 15);
  int src = ei[e];
  int dst = ei[N_EDGES_C + e];
  float4 v = reinterpret_cast<const float4*>(h + (long)src * CH)[q];
  float* ap = agg + (long)dst * CH + q * 4;
  atomicAdd(ap + 0, v.x);
  atomicAdd(ap + 1, v.y);
  atomicAdd(ap + 2, v.z);
  atomicAdd(ap + 3, v.w);
}

// ---------------- fused GIN MLP over one layer ----------------
// h <- relu((h+agg) @ W1 + b1) @ W2 + b2, in place.
// 4 waves/block; each wave computes a 16x64 row tile with v_wmma_f32_16x16x32_f16.
__global__ __launch_bounds__(128) void k_gin_mlp(float* __restrict__ h,
                                                 const float* __restrict__ agg,
                                                 const float* __restrict__ w1,
                                                 const float* __restrict__ pb1,
                                                 const float* __restrict__ w2,
                                                 const float* __restrict__ pb2) {
  __shared__ _Float16 w1t[CH * LSTR];        // W1^T in f16, padded rows
  __shared__ _Float16 w2t[CH * LSTR];        // W2^T in f16
  __shared__ _Float16 zt[4][16 * LSTR];      // per-wave z staging tile (f16)

  const int tid = threadIdx.x;
  // load both weight matrices transposed into LDS as f16
  for (int i = tid; i < CH * CH; i += 128) {
    int k = i >> 6, n = i & 63;                // w[k][n] (y = x @ W)
    w1t[n * LSTR + k] = (_Float16)w1[i];
    w2t[n * LSTR + k] = (_Float16)w2[i];
  }
  __syncthreads();

  const int wave = tid >> 5;
  const int lane = tid & 31;
  const int hl   = lane >> 4;                  // wave half (ISA A/C layout)
  const int m    = lane & 15;
  const long rowBase = (long)blockIdx.x * 64 + wave * 16;
  const long row = rowBase + m;

  // ---- A fragments for GEMM1: (h + agg) -> f16, WMMA 16x32 A layout ----
  // lane half 0: K = kc+0..15; half 1: K = kc+16..31
  const float4* hp = reinterpret_cast<const float4*>(h + row * CH);
  const float4* gp = reinterpret_cast<const float4*>(agg + row * CH);
  v16h afrag[2];
#pragma unroll
  for (int kc = 0; kc < 2; ++kc) {
#pragma unroll
    for (int gph = 0; gph < 2; ++gph) {
      int kbase = kc * 32 + hl * 16 + gph * 8;   // float index in the row
      float4 x0 = hp[(kbase >> 2) + 0], y0 = gp[(kbase >> 2) + 0];
      float4 x1 = hp[(kbase >> 2) + 1], y1 = gp[(kbase >> 2) + 1];
      int o = gph * 8;
      afrag[kc][o + 0] = (_Float16)(x0.x + y0.x);
      afrag[kc][o + 1] = (_Float16)(x0.y + y0.y);
      afrag[kc][o + 2] = (_Float16)(x0.z + y0.z);
      afrag[kc][o + 3] = (_Float16)(x0.w + y0.w);
      afrag[kc][o + 4] = (_Float16)(x1.x + y1.x);
      afrag[kc][o + 5] = (_Float16)(x1.y + y1.y);
      afrag[kc][o + 6] = (_Float16)(x1.z + y1.z);
      afrag[kc][o + 7] = (_Float16)(x1.w + y1.w);
    }
  }

  // ---- GEMM1 (K=64 as 2x K32) + bias + relu -> z tile in LDS ----
  _Float16* zrow = &zt[wave][0];
#pragma unroll
  for (int nt = 0; nt < 4; ++nt) {
    v8f acc = {0.f, 0.f, 0.f, 0.f, 0.f, 0.f, 0.f, 0.f};
#pragma unroll
    for (int kc = 0; kc < 2; ++kc) {
      // B layout: lane holds column n = lane&15; K = kc*32 + hl*16 + i
      const _Float16* bp = &w1t[(nt * 16 + m) * LSTR + kc * 32 + hl * 16];
      v16h bfrag;
#pragma unroll
      for (int i = 0; i < 16; ++i) bfrag[i] = bp[i];
      acc = wmma_f32_f16(afrag[kc], bfrag, acc);
    }
    float bias = pb1[nt * 16 + m];
#pragma unroll
    for (int r = 0; r < 8; ++r) {            // C layout: VGPR r -> row r + hl*8
      float v = acc[r] + bias;
      v = v > 0.f ? v : 0.f;
      zrow[(r + hl * 8) * LSTR + nt * 16 + m] = (_Float16)v;
    }
  }
  // wave-local RAW fence on the z tile (writes and reads are same-wave only)
  asm volatile("s_wait_dscnt 0" ::: "memory");

  // ---- A fragments for GEMM2 from the z tile ----
  v16h zfrag[2];
#pragma unroll
  for (int kc = 0; kc < 2; ++kc) {
    const _Float16* zp = zrow + m * LSTR + kc * 32 + hl * 16;
#pragma unroll
    for (int i = 0; i < 8; ++i) {
      zfrag[kc][i]     = zp[i];
      zfrag[kc][i + 8] = zp[8 + i];
    }
  }

  // ---- GEMM2 + bias -> h (in place) ----
#pragma unroll
  for (int nt = 0; nt < 4; ++nt) {
    v8f acc = {0.f, 0.f, 0.f, 0.f, 0.f, 0.f, 0.f, 0.f};
#pragma unroll
    for (int kc = 0; kc < 2; ++kc) {
      const _Float16* bp = &w2t[(nt * 16 + m) * LSTR + kc * 32 + hl * 16];
      v16h bfrag;
#pragma unroll
      for (int i = 0; i < 16; ++i) bfrag[i] = bp[i];
      acc = wmma_f32_f16(zfrag[kc], bfrag, acc);
    }
    float bias = pb2[nt * 16 + m];
#pragma unroll
    for (int r = 0; r < 8; ++r)
      h[(rowBase + r + hl * 8) * CH + nt * 16 + m] = acc[r] + bias;
  }
}

// ---------------- global_add_pool: g[batch[n]] += h[n] ----------------
__global__ __launch_bounds__(256) void k_pool(const int* __restrict__ batch,
                                              const float* __restrict__ h,
                                              float* __restrict__ g) {
  long gid = (long)blockIdx.x * blockDim.x + threadIdx.x;
  if (gid >= (long)N_NODES_C * 16) return;
  int node = (int)(gid >> 4), q = (int)(gid & 15);
  int b = batch[node];
  float4 v = reinterpret_cast<const float4*>(h + (long)node * CH)[q];
  float* gp = g + (long)b * CH + q * 4;
  atomicAdd(gp + 0, v.x);
  atomicAdd(gp + 1, v.y);
  atomicAdd(gp + 2, v.z);
  atomicAdd(gp + 3, v.w);
}

// ---------------- readout MLP 64->32->16->1 (4 MFLOP total) ----------------
__global__ __launch_bounds__(256) void k_readout(const float* __restrict__ g,
                                                 const float* __restrict__ mw1,
                                                 const float* __restrict__ mb1,
                                                 const float* __restrict__ mw2,
                                                 const float* __restrict__ mb2,
                                                 const float* __restrict__ mw3,
                                                 const float* __restrict__ mb3,
                                                 float* __restrict__ out) {
  int gi = blockIdx.x * blockDim.x + threadIdx.x;
  if (gi >= NUM_GRAPHS_C) return;
  float v[64];
#pragma unroll
  for (int k = 0; k < 64; ++k) v[k] = g[(long)gi * 64 + k];
  float t1[32];
  for (int j = 0; j < 32; ++j) {
    float s = mb1[j];
    for (int k = 0; k < 64; ++k) s += v[k] * mw1[k * 32 + j];
    t1[j] = s > 0.f ? s : 0.f;
  }
  float t2[16];
  for (int j = 0; j < 16; ++j) {
    float s = mb2[j];
    for (int k = 0; k < 32; ++k) s += t1[k] * mw2[k * 16 + j];
    t2[j] = s > 0.f ? s : 0.f;
  }
  float o = mb3[0];
  for (int k = 0; k < 16; ++k) o += t2[k] * mw3[k];
  out[gi] = o;
}

extern "C" void kernel_launch(void* const* d_in, const int* in_sizes, int n_in,
                              void* d_out, int out_size, void* d_ws, size_t ws_size,
                              hipStream_t stream) {
  (void)in_sizes; (void)n_in; (void)out_size; (void)ws_size;
  const int*   x        = (const int*)d_in[0];
  const int*   ei       = (const int*)d_in[1];   // [2, N_EDGES]
  // d_in[2] edge_attr and d_in[5] edge_emb are unused by the reference output
  const int*   batch    = (const int*)d_in[3];
  const float* node_emb = (const float*)d_in[4];
  const float* cw1      = (const float*)d_in[6];
  const float* cb1      = (const float*)d_in[7];
  const float* cw2      = (const float*)d_in[8];
  const float* cb2      = (const float*)d_in[9];
  const float* mw1      = (const float*)d_in[10];
  const float* mb1      = (const float*)d_in[11];
  const float* mw2      = (const float*)d_in[12];
  const float* mb2      = (const float*)d_in[13];
  const float* mw3      = (const float*)d_in[14];
  const float* mb3      = (const float*)d_in[15];

  float* h   = (float*)d_ws;                          // 200000*64 f32 (51.2 MB)
  float* agg = h + (size_t)N_NODES_C * CH;            // 200000*64 f32 (51.2 MB)
  float* g   = agg + (size_t)N_NODES_C * CH;          // 1024*64 f32
  float* out = (float*)d_out;

  const int thr = 256;
  const long nodeWork = (long)N_NODES_C * 16;
  const int  nodeBlocks = (int)((nodeWork + thr - 1) / thr);
  const long edgeWork = (long)N_EDGES_C * 16;
  const int  edgeBlocks = (int)((edgeWork + thr - 1) / thr);

  k_embed<<<nodeBlocks, thr, 0, stream>>>(x, node_emb, h);

  for (int l = 0; l < NUM_LAYERS_C; ++l) {
    k_zero<<<nodeBlocks, thr, 0, stream>>>((float4*)agg, (long)N_NODES_C * 16);
    k_scatter<<<edgeBlocks, thr, 0, stream>>>(ei, h, agg);
    k_gin_mlp<<<N_NODES_C / 64, 128, 0, stream>>>(
        h, agg,
        cw1 + (size_t)l * CH * CH, cb1 + (size_t)l * CH,
        cw2 + (size_t)l * CH * CH, cb2 + (size_t)l * CH);
  }

  k_zero<<<(NUM_GRAPHS_C * 16 + thr - 1) / thr, thr, 0, stream>>>(
      (float4*)g, (long)NUM_GRAPHS_C * 16);
  k_pool<<<nodeBlocks, thr, 0, stream>>>(batch, h, g);
  k_readout<<<(NUM_GRAPHS_C + thr - 1) / thr, thr, 0, stream>>>(
      g, mw1, mb1, mw2, mb2, mw3, mb3, out);
}
